// GGNN_40037685133968
// MI455X (gfx1250) — matrix operand order repaired and checked by reading, not compiled, not run on hardware
//
#include <hip/hip_runtime.h>
#include <hip/hip_bf16.h>
#include <math.h>

typedef __bf16 bf16;
typedef __attribute__((ext_vector_type(16))) __bf16 v16bf;
typedef __attribute__((ext_vector_type(8)))  __bf16 v8bf;
typedef __attribute__((ext_vector_type(8)))  float  v8f;

#define B_   128
#define N_   64
#define F_   64
#define E_   16
#define H_   128
#define OUT_ 12
#define M_   (B_*N_)          /* 8192 rows of node state */
#define HF_  (H_+F_)          /* 192 */

__device__ __forceinline__ float sigmoidf_(float x){ return 1.0f/(1.0f+__expf(-x)); }

#define WMMA_BF16(A,Bm,C) __builtin_amdgcn_wmma_f32_16x16x32_bf16(false,(A),false,(Bm),(short)0,(C),false,false)

// ---------------- prep kernels ----------------
__global__ void cvt_f32_bf16(const float* __restrict__ s, bf16* __restrict__ d, int n){
  int i = blockIdx.x*blockDim.x + threadIdx.x;
  if (i < n) d[i] = (bf16)s[i];
}

__global__ void init_node(const float* __restrict__ nin, float* __restrict__ nf,
                          bf16* __restrict__ nb){
  int i = blockIdx.x*blockDim.x + threadIdx.x;      // M_*H_ total
  int row = i >> 7, h = i & (H_-1);
  float v = (h < F_) ? nin[row*F_ + h] : 0.0f;
  nf[i] = v; nb[i] = (bf16)v;
}

__global__ void zero_f32(float* __restrict__ p, int n){
  int i = blockIdx.x*blockDim.x + threadIdx.x;
  if (i < n) p[i] = 0.0f;
}

__global__ void build_xcat(const float* __restrict__ nxf, const float* __restrict__ nin,
                           bf16* __restrict__ xcat){
  int i = blockIdx.x*blockDim.x + threadIdx.x;      // M_*HF_ total
  int row = i / HF_, c = i - row*HF_;
  float v = (c < H_) ? nxf[(size_t)row*H_ + c] : nin[(size_t)row*F_ + (c - H_)];
  xcat[i] = (bf16)v;
}

// ---------------- edge embedding: tanh(edge @ We + be) -> bf16, b128 stores -------
// Block: 128 threads = 8 edges x 16 lane-groups; each thread emits 8 channels.
__global__ void edge_embed(const float* __restrict__ ein, const float* __restrict__ We,
                           const float* __restrict__ be, bf16* __restrict__ eout){
  __shared__ float sW[E_*H_];
  __shared__ float sx[8][E_];
  int t = threadIdx.x;                              // 128
  #pragma unroll
  for (int e = 0; e < E_; ++e) sW[e*H_ + t] = We[e*H_ + t];
  int r  = t >> 4;                                  // edge sub-index 0..7
  int l  = t & 15;
  int h0 = l * 8;
  size_t base = (size_t)blockIdx.x * 8;
  sx[r][l] = ein[(base + r)*E_ + l];                // 16 features per edge, one per lane
  __syncthreads();
  float acc[8];
  #pragma unroll
  for (int c = 0; c < 8; ++c) acc[c] = be[h0 + c];
  #pragma unroll
  for (int e = 0; e < E_; ++e){
    float x = sx[r][e];
    #pragma unroll
    for (int c = 0; c < 8; ++c) acc[c] += x * sW[e*H_ + h0 + c];
  }
  v8bf out;
  #pragma unroll
  for (int c = 0; c < 8; ++c) out[c] = (bf16)tanhf(acc[c]);
  *(v8bf*)(eout + (base + r)*H_ + h0) = out;
}

// ---------------- msg[bi,h] = sum_j nx[b,j,h]*edge[bi,j,h] : b128 streaming -------
// Block: 128 threads = 8 rows (b,i) x 16 lane-groups; thread owns 8 channels.
// One 16B edge load + one 16B nx load per neighbor j -> 8x fewer vmem ops.
__global__ void msg_kernel(const bf16* __restrict__ edge, const bf16* __restrict__ nx,
                           bf16* __restrict__ msg){
  int t  = threadIdx.x;
  int r  = t >> 4;                                  // sub-row 0..7
  int l  = t & 15;
  int h0 = l * 8;
  size_t bi = (size_t)blockIdx.x*8 + r;             // b*N_ + i  (same b for all 8 rows)
  int b = (int)(bi >> 6);
  const bf16* ep = edge + bi*(size_t)(N_*H_) + h0;
  const bf16* np = nx   + (size_t)b*(N_*H_) + h0;
  float acc[8] = {0.f,0.f,0.f,0.f,0.f,0.f,0.f,0.f};
  #pragma unroll 4
  for (int j = 0; j < N_; ++j){
    v8bf ev = *(const v8bf*)(ep + (size_t)j*H_);
    v8bf nv = *(const v8bf*)(np + (size_t)j*H_);
    #pragma unroll
    for (int c = 0; c < 8; ++c) acc[c] += (float)ev[c] * (float)nv[c];
  }
  v8bf out;
  #pragma unroll
  for (int c = 0; c < 8; ++c) out[c] = (bf16)acc[c];
  *(v8bf*)(msg + bi*(size_t)H_ + h0) = out;
}

// ---------------- fused GRU step via WMMA bf16 ----------------
// hnew = (1-z)*tanh(xh + r*hh) + z*h0 ; z=sig(xz+hz), r=sig(xr+hr)
// Block: 16 rows x 128 cols, 8 waves (wave w -> col tile n0=16w), K=128 (4 k-steps).
__global__ void gru_kernel(const bf16* __restrict__ Xb, const bf16* __restrict__ H0b,
                           const float* __restrict__ H0f,
                           const bf16* __restrict__ Wxb, const bf16* __restrict__ Whb,
                           const float* __restrict__ bx, const float* __restrict__ bh,
                           float* __restrict__ Hf, bf16* __restrict__ Hb){
  const int m0   = blockIdx.x * 16;
  const int wave = threadIdx.x >> 5;
  const int lane = threadIdx.x & 31;
  const int n0   = wave * 16;
  const int arow = m0 + (lane & 15);                // A frag: lane = row, k-half by lane grp
  const int kh   = (lane >> 4) * 16;

  v16bf ax[4], ah[4];
  #pragma unroll
  for (int ks = 0; ks < 4; ++ks){
    int k = ks*32 + kh;
    ax[ks] = *(const v16bf*)(Xb  + (size_t)arow*H_ + k);
    ah[ks] = *(const v16bf*)(H0b + (size_t)arow*H_ + k);
  }
  v8f xz{}, xr{}, xh{}, hz{}, hr{}, hh{};
  #pragma unroll
  for (int ks = 0; ks < 4; ++ks){
    int k = ks*32 + kh;                             // B frag: lane = N col, 16 K values
    const bf16* wx = Wxb + (size_t)k*(3*H_) + (lane & 15);
    const bf16* wh = Whb + (size_t)k*(3*H_) + (lane & 15);
    v16bf bz, br, bH, cz, cr, cH;
    #pragma unroll
    for (int s = 0; s < 16; ++s){
      bz[s] = wx[(size_t)s*(3*H_) + n0];
      br[s] = wx[(size_t)s*(3*H_) + n0 + H_];
      bH[s] = wx[(size_t)s*(3*H_) + n0 + 2*H_];
      cz[s] = wh[(size_t)s*(3*H_) + n0];
      cr[s] = wh[(size_t)s*(3*H_) + n0 + H_];
      cH[s] = wh[(size_t)s*(3*H_) + n0 + 2*H_];
    }
    xz = WMMA_BF16(ax[ks], bz, xz);
    xr = WMMA_BF16(ax[ks], br, xr);
    xh = WMMA_BF16(ax[ks], bH, xh);
    hz = WMMA_BF16(ah[ks], cz, hz);
    hr = WMMA_BF16(ah[ks], cr, hr);
    hh = WMMA_BF16(ah[ks], cH, hh);
  }
  const int   nc  = n0 + (lane & 15);               // C/D frag: col = lane%16, row = g+8*(lane/16)
  const float bxz = bx[nc], bxr = bx[nc+H_], bxh = bx[nc+2*H_];
  const float bhz = bh[nc], bhr = bh[nc+H_], bhh = bh[nc+2*H_];
  const int rbase = m0 + 8*(lane>>4);
  #pragma unroll
  for (int g = 0; g < 8; ++g){
    int row = rbase + g;
    float z  = sigmoidf_(xz[g]+bxz + hz[g]+bhz);
    float r  = sigmoidf_(xr[g]+bxr + hr[g]+bhr);
    float hc = tanhf(xh[g]+bxh + r*(hh[g]+bhh));
    float h0 = H0f[(size_t)row*H_ + nc];
    float hn = (1.0f - z)*hc + z*h0;
    Hf[(size_t)row*H_ + nc] = hn;
    Hb[(size_t)row*H_ + nc] = (bf16)hn;
  }
}

// ---------------- readout: accum[b,h] += sum_rows sig(x@W1+b1)*tanh(x@W2+b2) ------
__global__ void readout_kernel(const bf16* __restrict__ xcat, const bf16* __restrict__ W1b,
                               const bf16* __restrict__ W2b, const float* __restrict__ b1,
                               const float* __restrict__ b2, float* __restrict__ accum){
  const int m0   = blockIdx.x * 16;
  const int wave = threadIdx.x >> 5;
  const int lane = threadIdx.x & 31;
  const int n0   = wave * 16;
  const int arow = m0 + (lane & 15);
  const int kh   = (lane >> 4) * 16;
  v8f a1{}, a2{};
  #pragma unroll
  for (int ks = 0; ks < 6; ++ks){                   // K = 192
    int ka = ks*32 + kh;
    v16bf av = *(const v16bf*)(xcat + (size_t)arow*HF_ + ka);
    int kb = ks*32 + kh;
    v16bf w1, w2;
    #pragma unroll
    for (int s = 0; s < 16; ++s){
      w1[s] = W1b[(size_t)(kb + s)*H_ + n0 + (lane & 15)];
      w2[s] = W2b[(size_t)(kb + s)*H_ + n0 + (lane & 15)];
    }
    a1 = WMMA_BF16(av, w1, a1);
    a2 = WMMA_BF16(av, w2, a2);
  }
  const int nc = n0 + (lane & 15);
  const float bb1 = b1[nc], bb2 = b2[nc];
  float part = 0.f;
  #pragma unroll
  for (int g = 0; g < 8; ++g)
    part += sigmoidf_(a1[g] + bb1) * tanhf(a2[g] + bb2);
  int b = m0 >> 6;                                  // 16-row tile lies within one batch
  atomicAdd(&accum[b*H_ + nc], part);
}

__global__ void final_kernel(const float* __restrict__ accum, const float* __restrict__ Wo,
                             const float* __restrict__ bo, float* __restrict__ out){
  int b = threadIdx.x;                              // 128 threads, one batch each
  if (b >= B_) return;
  float s[OUT_];
  #pragma unroll
  for (int o = 0; o < OUT_; ++o) s[o] = bo[o];
  for (int h = 0; h < H_; ++h){
    float g = tanhf(accum[b*H_ + h]);
    #pragma unroll
    for (int o = 0; o < OUT_; ++o) s[o] += g * Wo[h*OUT_ + o];
  }
  #pragma unroll
  for (int o = 0; o < OUT_; ++o) out[b*OUT_ + o] = sigmoidf_(s[o]);
}

// ---------------- host side ----------------
extern "C" void kernel_launch(void* const* d_in, const int* in_sizes, int n_in,
                              void* d_out, int out_size, void* d_ws, size_t ws_size,
                              hipStream_t stream){
  (void)in_sizes; (void)n_in; (void)out_size; (void)ws_size;
  const float* node_in = (const float*)d_in[0];
  const float* edge_in = (const float*)d_in[1];
  const float* We = (const float*)d_in[2];
  const float* be = (const float*)d_in[3];
  const float* W1 = (const float*)d_in[4];
  const float* b1 = (const float*)d_in[5];
  const float* W2 = (const float*)d_in[6];
  const float* b2 = (const float*)d_in[7];
  const float* Wo = (const float*)d_in[8];
  const float* bo = (const float*)d_in[9];
  const float* gWx[3] = {(const float*)d_in[10], (const float*)d_in[14], (const float*)d_in[18]};
  const float* gWh[3] = {(const float*)d_in[11], (const float*)d_in[15], (const float*)d_in[19]};
  const float* gbx[3] = {(const float*)d_in[12], (const float*)d_in[16], (const float*)d_in[20]};
  const float* gbh[3] = {(const float*)d_in[13], (const float*)d_in[17], (const float*)d_in[21]};

  char* ws = (char*)d_ws;
  size_t off = 0;
  auto carve = [&](size_t bytes)->char*{
    char* p = ws + off; off += (bytes + 255) & ~(size_t)255; return p;
  };
  bf16*  edge_bf = (bf16*) carve((size_t)B_*N_*N_*H_*2);   // 128 MB — fits 192MB L2
  float* nxAf    = (float*)carve((size_t)M_*H_*4);
  bf16*  nxAb    = (bf16*) carve((size_t)M_*H_*2);
  float* nxBf    = (float*)carve((size_t)M_*H_*4);
  bf16*  nxBb    = (bf16*) carve((size_t)M_*H_*2);
  float* t1f     = (float*)carve((size_t)M_*H_*4);
  bf16*  t1b     = (bf16*) carve((size_t)M_*H_*2);
  float* t2f     = (float*)carve((size_t)M_*H_*4);
  bf16*  t2b     = (bf16*) carve((size_t)M_*H_*2);
  bf16*  msgb    = (bf16*) carve((size_t)M_*H_*2);
  bf16*  Wxb[3], *Whb[3];
  for (int i = 0; i < 3; ++i){
    Wxb[i] = (bf16*)carve((size_t)H_*3*H_*2);
    Whb[i] = (bf16*)carve((size_t)H_*3*H_*2);
  }
  bf16*  W1b  = (bf16*) carve((size_t)HF_*H_*2);
  bf16*  W2b  = (bf16*) carve((size_t)HF_*H_*2);
  bf16*  xcat = (bf16*) carve((size_t)M_*HF_*2);
  float* accum= (float*)carve((size_t)B_*H_*4);

  // ---- weight conversion to bf16 ----
  const int nW = H_*3*H_;                           // 49152
  for (int i = 0; i < 3; ++i){
    cvt_f32_bf16<<<(nW+255)/256, 256, 0, stream>>>(gWx[i], Wxb[i], nW);
    cvt_f32_bf16<<<(nW+255)/256, 256, 0, stream>>>(gWh[i], Whb[i], nW);
  }
  const int nW12 = HF_*H_;                          // 24576
  cvt_f32_bf16<<<(nW12+255)/256, 256, 0, stream>>>(W1, W1b, nW12);
  cvt_f32_bf16<<<(nW12+255)/256, 256, 0, stream>>>(W2, W2b, nW12);

  // ---- node state init + edge embedding ----
  init_node<<<(M_*H_)/256, 256, 0, stream>>>(node_in, nxAf, nxAb);
  edge_embed<<<(B_*N_*N_)/8, 128, 0, stream>>>(edge_in, We, be, edge_bf);

  // ---- 10 propagation steps ----
  float* curf = nxAf; bf16* curb = nxAb;
  float* nxtf = nxBf; bf16* nxtb = nxBb;
  for (int t = 0; t < 10; ++t){
    msg_kernel<<<M_/8, 128, 0, stream>>>(edge_bf, curb, msgb);
    gru_kernel<<<M_/16, 256, 0, stream>>>(msgb, curb, curf, Wxb[0], Whb[0], gbx[0], gbh[0], t1f, t1b);
    gru_kernel<<<M_/16, 256, 0, stream>>>(t1b,  curb, curf, Wxb[1], Whb[1], gbx[1], gbh[1], t2f, t2b);
    gru_kernel<<<M_/16, 256, 0, stream>>>(t2b,  curb, curf, Wxb[2], Whb[2], gbx[2], gbh[2], nxtf, nxtb);
    float* tf = curf; curf = nxtf; nxtf = tf;
    bf16*  tb = curb; curb = nxtb; nxtb = tb;
  }

  // ---- readout ----
  build_xcat<<<(M_*HF_)/256, 256, 0, stream>>>(curf, node_in, xcat);
  zero_f32<<<(B_*H_+255)/256, 256, 0, stream>>>(accum, B_*H_);
  readout_kernel<<<M_/16, 256, 0, stream>>>(xcat, W1b, W2b, b1, b2, accum);
  final_kernel<<<1, 128, 0, stream>>>(accum, Wo, bo, (float*)d_out);
}
